// meta_LSTMModel_51127290691978
// MI455X (gfx1250) — matrix operand order repaired and checked by reading, not compiled, not run on hardware
//
#include <hip/hip_runtime.h>

// ---------------------------------------------------------------------------
// CDNA5 (gfx1250) bf16-WMMA persistent-scan implementation of the 2-layer
// meta-LSTM.
//  - All GEMMs use v_wmma_f32_16x16x32_bf16 (wave32, B=32 -> 2 m-tiles).
//  - Weights are packed once per call into per-lane WMMA B-tile layout and
//    stay resident in the 192MB L2 across the 400-step scan (~82MB bf16).
//  - The entire 2-layer x 400-step recurrence runs in ONE persistent kernel;
//    phases are separated by device-wide sense-reversing barriers built from
//    agent-scope atomics + acquire/release fences, with s_sleep in the spin.
//  - All hot-path memory accesses are explicitly addrspace(1) (global_load/
//    global_store, never flat) so no DScnt coupling and no aperture checks.
// ---------------------------------------------------------------------------

typedef __attribute__((ext_vector_type(16))) __bf16 bf16x16;
typedef __attribute__((ext_vector_type(8)))  float  f32x8;
typedef unsigned short u16;
typedef unsigned int   u32;
typedef __attribute__((ext_vector_type(4)))  u32    u32x4;   // builtin 128-bit vec

#define BATCH 32
#define TT    400
#define DIN   161
#define DINP  192      // padded layer-0 input width (mult of 32)
#define DH    1024
#define DHH   512
#define NCOUT 161
#define NBLK  68       // persistent blocks (x 8 waves = 544 waves)

// ---------------- global address-space helpers ----------------

typedef __attribute__((address_space(1))) const u32x4* gc_v4p;
typedef __attribute__((address_space(1))) u32x4*       g_v4p;

static __device__ __forceinline__ float gldf(const float* p) {
  return *(__attribute__((address_space(1))) const float*)p;
}
static __device__ __forceinline__ void gstf(float* p, float v) {
  *(__attribute__((address_space(1))) float*)p = v;
}
static __device__ __forceinline__ void gstu(u16* p, u16 v) {
  *(__attribute__((address_space(1))) u16*)p = v;
}
static __device__ __forceinline__ void gst32(u32* p, u32 v) {
  *(__attribute__((address_space(1))) u32*)p = v;
}

// ---------------- device helpers ----------------

static __device__ __forceinline__ u16 f2bf(float f) {
  union { float f; u32 u; } v; v.f = f;
  u32 u = v.u;
  u32 r = u + 0x7fffu + ((u >> 16) & 1u);   // RNE
  return (u16)(r >> 16);
}
static __device__ __forceinline__ float sigm(float x) {
  return 1.0f / (1.0f + __expf(-x));
}
static __device__ __forceinline__ float tanh_f(float x) {
  float a = fabsf(x);
  float e = __expf(-2.0f * a);
  float r = (1.0f - e) / (1.0f + e);
  return x < 0.0f ? -r : r;
}
static __device__ __forceinline__ f32x8 zero8() {
  f32x8 v; for (int i = 0; i < 8; ++i) v[i] = 0.0f; return v;
}
static __device__ __forceinline__ f32x8 wmma_bf(bf16x16 a, bf16x16 b, f32x8 c) {
  return __builtin_amdgcn_wmma_f32_16x16x32_bf16(false, a, false, b, (short)0, c, false, false);
}
// A fragment (16x32 bf16): lane l holds row m=l&15, K = {koff..koff+7, koff+16..koff+23},
// koff = (l>=16)*8.  p points at A[m][k0+koff]; elems 0..7 at p, 8..15 at p+16.
static __device__ __forceinline__ bf16x16 fragA(const u16* p) {
  union { bf16x16 v; u32x4 q[2]; } u;
  gc_v4p gp = (gc_v4p)(const u32x4*)p;
  u.q[0] = gp[0];
  u.q[1] = gp[2];   // +16 bf16 = +32 bytes
  return u.v;
}
// B fragment from packed tile: 16 contiguous bf16 per lane.
static __device__ __forceinline__ bf16x16 fragB(const u16* tile, int lane) {
  union { bf16x16 v; u32x4 q[2]; } u;
  gc_v4p gp = (gc_v4p)(const u32x4*)(tile + lane * 16);
  u.q[0] = gp[0];
  u.q[1] = gp[1];
  return u.v;
}

// device-wide sense-reversing barrier (one arrival per block)
static __device__ __forceinline__ void gbar(u32* cnt, u32* epoch_g, u32& eloc, int nblocks) {
  __syncthreads();
  if (threadIdx.x == 0) {
    __builtin_amdgcn_fence(__ATOMIC_RELEASE, "agent");
    u32 prev = __hip_atomic_fetch_add(cnt, 1u, __ATOMIC_RELAXED, __HIP_MEMORY_SCOPE_AGENT);
    u32 target = eloc + 1u;
    if (prev == (u32)(nblocks - 1)) {
      __hip_atomic_store(cnt, 0u, __ATOMIC_RELAXED, __HIP_MEMORY_SCOPE_AGENT);
      __hip_atomic_store(epoch_g, target, __ATOMIC_RELEASE, __HIP_MEMORY_SCOPE_AGENT);
    } else {
      while (__hip_atomic_load(epoch_g, __ATOMIC_ACQUIRE, __HIP_MEMORY_SCOPE_AGENT) < target)
        __builtin_amdgcn_s_sleep(2);
    }
    __builtin_amdgcn_fence(__ATOMIC_ACQUIRE, "agent");
  }
  __syncthreads();
  eloc += 1u;
}

// ---------------- setup kernels ----------------

__global__ void k_cvt_x(const float* __restrict__ x, u16* __restrict__ xbf, int total) {
  int i = blockIdx.x * 256 + threadIdx.x;
  if (i >= total) return;
  int k  = i % DINP;
  int bt = i / DINP;
  xbf[i] = (k < DIN) ? f2bf(x[(size_t)bt * DIN + k]) : (u16)0;
}

__global__ void k_zero(u32* p, int n) {
  int i = blockIdx.x * 256 + threadIdx.x;
  if (i < n) p[i] = 0u;
}

__global__ void k_zbias(const float* __restrict__ bzx, const float* __restrict__ bzh,
                        float* __restrict__ zb) {
  int i = blockIdx.x * 256 + threadIdx.x;
  if (i >= 6144) return;
  zb[i] = (i < 2048) ? bzx[i] : (i < 4096 ? bzh[i - 2048] : 0.0f);
}

// Pack fp32 row-major W block into WMMA-B tiles (bf16).
// B element e of lane l in tile (kt,nt): k = kt*32 + (l>=16?8:0) + (e<8 ? e : e+8), n = nt*16 + (l&15).
// Packed tile layout: [nt][kt][lane][e], 512 bf16 (1KB) per tile.
__global__ __launch_bounds__(32)
void k_pack_b(const float* __restrict__ W, int ldw, int src_k0, int src_n0,
              int valid_k, int valid_n, u16* __restrict__ out,
              int out_kt0, int out_nt0, int out_Ktiles, int nkt) {
  int tile = blockIdx.x;
  int kt = tile % nkt;
  int nt = tile / nkt;
  int lane = threadIdx.x;
  int kloc0 = kt * 32 + ((lane >> 4) * 8);
  int nloc  = nt * 16 + (lane & 15);
  alignas(16) u16 vals[16];
#pragma unroll
  for (int e = 0; e < 16; ++e) {
    int kloc = kloc0 + e + (e >= 8 ? 8 : 0);
    float f = 0.0f;
    if (kloc < valid_k && nloc < valid_n)
      f = W[(size_t)(src_k0 + kloc) * ldw + (src_n0 + nloc)];
    vals[e] = f2bf(f);
  }
  size_t tix = (size_t)(out_nt0 + nt) * out_Ktiles + (out_kt0 + kt);
  u16* o = out + tix * 512 + lane * 16;
  g_v4p go = (g_v4p)(u32x4*)o;
  const u32x4* vv = (const u32x4*)vals;
  go[0] = vv[0];
  go[1] = vv[1];
}

// ---------------- persistent scan kernel ----------------

struct LayerW {
  const u16* xsrc;                 // layer input (bf16, padded): xbf or h1s
  const u16 *whg, *wxp, *whp, *wz, *wd;
  const float *hb, *b0, *zbias;
  u16* hstore;                     // per-step h output (bf16, (B,T,DH))
  int d_pad, kh, kx, xrs;
};

struct PArgs {
  LayerW L[2];
  u16 *h0, *h1, *hh0, *hh1;
  float *c0, *c1, *hc0, *hc1;
  float *ax, *ah;
  u16* zbuf;
  const u16* wlinp;
  const float* blin;
  float* out;
  u32 *cnt, *epoch;
};

// plain 32 x (2*16) GEMM, fp32 store (ax / ah), out stride 4096
static __device__ __forceinline__
void gemm2_store_f32(const u16* Abase, int arstride, size_t aoff, int ktiles,
                     const u16* Wp, int wKtiles, int nt0, float* outp, int lane) {
  f32x8 acc[2][2];
  acc[0][0] = zero8(); acc[0][1] = zero8(); acc[1][0] = zero8(); acc[1][1] = zero8();
  int mA = lane & 15;
  int koff = (lane >> 4) * 8;
  const u16* r0 = Abase + (size_t)mA * arstride + aoff + koff;
  const u16* r1 = Abase + (size_t)(mA + 16) * arstride + aoff + koff;
  for (int kt = 0; kt < ktiles; ++kt) {
    int kb = kt * 32;
    bf16x16 a0 = fragA(r0 + kb);
    bf16x16 a1 = fragA(r1 + kb);
#pragma unroll
    for (int nn = 0; nn < 2; ++nn) {
      bf16x16 b = fragB(Wp + ((size_t)(nt0 + nn) * wKtiles + kt) * 512, lane);
      acc[nn][0] = wmma_bf(a0, b, acc[nn][0]);
      acc[nn][1] = wmma_bf(a1, b, acc[nn][1]);
    }
  }
  int n = lane & 15, mro = (lane >> 4) * 8;
#pragma unroll
  for (int nn = 0; nn < 2; ++nn) {
    int col = (nt0 + nn) * 16 + n;
#pragma unroll
    for (int mt = 0; mt < 2; ++mt)
#pragma unroll
      for (int r = 0; r < 8; ++r)
        gstf(outp + (size_t)(mt * 16 + mro + r) * 4096 + col, acc[nn][mt][r]);
  }
}

// accumulate one k-tile into the 4-gate hg accumulators
static __device__ __forceinline__
void hg_accum(f32x8 (&acc)[4][2], const u16* r0, const u16* r1,
              const u16* whg, int kh, int j, int ktglob, int lane) {
  bf16x16 a0 = fragA(r0);
  bf16x16 a1 = fragA(r1);
#pragma unroll
  for (int g = 0; g < 4; ++g) {
    bf16x16 b = fragB(whg + ((size_t)(g * 32 + j) * kh + ktglob) * 512, lane);
    acc[g][0] = wmma_bf(a0, b, acc[g][0]);
    acc[g][1] = wmma_bf(a1, b, acc[g][1]);
  }
}

// phase 1 task: [0,32) hg GEMM + meta cell; [32,160) ax; [160,288) ah
static __device__ void stage1_task(int task, int lane, const LayerW& W, int t,
                                   const u16* h_cur, const u16* hh_cur,
                                   const float* hc_cur, float* hc_nxt, u16* hh_nxt,
                                   float* ax, float* ah) {
  if (task >= 160) {
    gemm2_store_f32(h_cur, DH, 0, 32, W.whp, 32, (task - 160) * 2, ah, lane);
    return;
  }
  if (task >= 32) {
    gemm2_store_f32(W.xsrc, W.xrs, (size_t)t * W.d_pad, W.kx, W.wxp, W.kx,
                    (task - 32) * 2, ax, lane);
    return;
  }
  int j = task;  // 16-column slice of HH=512
  f32x8 acc[4][2];
#pragma unroll
  for (int g = 0; g < 4; ++g) { acc[g][0] = zero8(); acc[g][1] = zero8(); }
  int mA = lane & 15;
  int koff = (lane >> 4) * 8;
  // segment 0: x rows (branch-free uniform loop)
  {
    const u16* r0 = W.xsrc + (size_t)mA * W.xrs + (size_t)t * W.d_pad + koff;
    const u16* r1 = W.xsrc + (size_t)(mA + 16) * W.xrs + (size_t)t * W.d_pad + koff;
    for (int kt = 0; kt < W.kx; ++kt)
      hg_accum(acc, r0 + kt * 32, r1 + kt * 32, W.whg, W.kh, j, kt, lane);
  }
  // segment 1: h rows (K = 1024)
  {
    const u16* r0 = h_cur + mA * DH + koff;
    const u16* r1 = h_cur + (mA + 16) * DH + koff;
#pragma unroll 4
    for (int k2 = 0; k2 < 32; ++k2)
      hg_accum(acc, r0 + k2 * 32, r1 + k2 * 32, W.whg, W.kh, j, W.kx + k2, lane);
  }
  // segment 2: hh rows (K = 512)
  {
    const u16* r0 = hh_cur + mA * DHH + koff;
    const u16* r1 = hh_cur + (mA + 16) * DHH + koff;
#pragma unroll 4
    for (int k2 = 0; k2 < 16; ++k2)
      hg_accum(acc, r0 + k2 * 32, r1 + k2 * 32, W.whg, W.kh, j, W.kx + 32 + k2, lane);
  }
  // fused meta-LSTM cell update on this 16-column slice
  int n = lane & 15, mro = (lane >> 4) * 8;
  int col = j * 16 + n;
  float bi = gldf(W.hb + col), bf_ = gldf(W.hb + 512 + col);
  float bn = gldf(W.hb + 1024 + col), bo = gldf(W.hb + 1536 + col);
#pragma unroll
  for (int mt = 0; mt < 2; ++mt)
#pragma unroll
    for (int r = 0; r < 8; ++r) {
      int m = mt * 16 + mro + r;
      float iv = acc[0][mt][r] + bi;
      float fv = acc[1][mt][r] + bf_;
      float nv = acc[2][mt][r] + bn;
      float ov = acc[3][mt][r] + bo;
      float hco = gldf(hc_cur + m * DHH + col);
      float hcn = sigm(fv) * hco + sigm(iv) * tanh_f(nv);
      gstf(hc_nxt + m * DHH + col, hcn);
      gstu(hh_nxt + m * DHH + col, f2bf(sigm(ov) * tanh_f(hcn)));
    }
}

// phase 2 task: z = hh @ [Wzx|Wzh|Wzb] (+biases), bf16 out (32 x 6144)
static __device__ void stage2_task(int task, int lane, const LayerW& W,
                                   const u16* hh, u16* z) {
  int nt0 = task * 2;
  f32x8 acc[2][2];
  acc[0][0] = zero8(); acc[0][1] = zero8(); acc[1][0] = zero8(); acc[1][1] = zero8();
  int mA = lane & 15;
  int koff = (lane >> 4) * 8;
  const u16* r0 = hh + mA * DHH + koff;
  const u16* r1 = hh + (mA + 16) * DHH + koff;
#pragma unroll
  for (int kt = 0; kt < 16; ++kt) {
    bf16x16 a0 = fragA(r0 + kt * 32);
    bf16x16 a1 = fragA(r1 + kt * 32);
#pragma unroll
    for (int nn = 0; nn < 2; ++nn) {
      bf16x16 b = fragB(W.wz + ((size_t)(nt0 + nn) * 16 + kt) * 512, lane);
      acc[nn][0] = wmma_bf(a0, b, acc[nn][0]);
      acc[nn][1] = wmma_bf(a1, b, acc[nn][1]);
    }
  }
  int n = lane & 15, mro = (lane >> 4) * 8;
#pragma unroll
  for (int nn = 0; nn < 2; ++nn) {
    int col = (nt0 + nn) * 16 + n;
    float zb = gldf(W.zbias + col);
#pragma unroll
    for (int mt = 0; mt < 2; ++mt)
#pragma unroll
      for (int r = 0; r < 8; ++r) {
        int m = mt * 16 + mro + r;
        gstu(z + (size_t)m * 6144 + col, f2bf(acc[nn][mt][r] + zb));
      }
  }
}

// phase 3 task: 12 einsum GEMMs (dx/dh/db x 4 gates) fused with main LSTM cell
static __device__ void stage3_task(int task, int lane, const LayerW& W,
                                   const u16* z, const float* ax, const float* ah,
                                   const float* c_cur, float* c_nxt,
                                   u16* h_nxt, int t) {
  int j = task;  // 16-column slice of H=1024
  f32x8 acc[4][3][2];
#pragma unroll
  for (int g = 0; g < 4; ++g)
#pragma unroll
    for (int s = 0; s < 3; ++s) { acc[g][s][0] = zero8(); acc[g][s][1] = zero8(); }
  int mA = lane & 15;
  int koff = (lane >> 4) * 8;
#pragma unroll
  for (int s = 0; s < 3; ++s)
#pragma unroll
    for (int g = 0; g < 4; ++g) {
      const u16* r0 = z + (size_t)mA * 6144 + s * 2048 + g * 512 + koff;
      const u16* r1 = z + (size_t)(mA + 16) * 6144 + s * 2048 + g * 512 + koff;
      const u16* wmat = W.wd + (size_t)(s * 4 + g) * 524288;  // 512x1024 packed
#pragma unroll
      for (int kt = 0; kt < 16; ++kt) {
        bf16x16 a0 = fragA(r0 + kt * 32);
        bf16x16 a1 = fragA(r1 + kt * 32);
        bf16x16 b = fragB(wmat + ((size_t)(j * 16 + kt)) * 512, lane);
        acc[g][s][0] = wmma_bf(a0, b, acc[g][s][0]);
        acc[g][s][1] = wmma_bf(a1, b, acc[g][s][1]);
      }
    }
  int n = lane & 15, mro = (lane >> 4) * 8;
  int col = j * 16 + n;
  float bb[4];
#pragma unroll
  for (int g = 0; g < 4; ++g) bb[g] = gldf(W.b0 + g * DH + col);
#pragma unroll
  for (int mt = 0; mt < 2; ++mt)
#pragma unroll
    for (int r = 0; r < 8; ++r) {
      int m = mt * 16 + mro + r;
      float pre[4];
#pragma unroll
      for (int g = 0; g < 4; ++g)
        pre[g] = acc[g][0][mt][r] * gldf(ax + (size_t)m * 4096 + g * DH + col)
               + acc[g][1][mt][r] * gldf(ah + (size_t)m * 4096 + g * DH + col)
               + acc[g][2][mt][r] + bb[g];
      float cn = sigm(pre[1]) * gldf(c_cur + m * DH + col) + sigm(pre[0]) * tanh_f(pre[2]);
      gstf(c_nxt + m * DH + col, cn);
      u16 hb16 = f2bf(sigm(pre[3]) * tanh_f(cn));
      gstu(h_nxt + m * DH + col, hb16);
      gstu(W.hstore + ((size_t)m * TT + t) * DH + col, hb16);
    }
}

// final task: one 16-row slice of out = sigmoid(h2 @ Wlin + blin)
static __device__ void final_task(int task, int lane, const u16* h2,
                                  const u16* wlin, const float* blin, float* out) {
  int mrow0 = task * 16;
  f32x8 acc[11];
#pragma unroll
  for (int nv = 0; nv < 11; ++nv) acc[nv] = zero8();
  int mA = lane & 15;
  int koff = (lane >> 4) * 8;
  const u16* row = h2 + (size_t)(mrow0 + mA) * DH + koff;
  for (int kt = 0; kt < 32; ++kt) {
    bf16x16 a = fragA(row + kt * 32);
#pragma unroll
    for (int nv = 0; nv < 11; ++nv) {
      bf16x16 b = fragB(wlin + ((size_t)(nv * 32 + kt)) * 512, lane);
      acc[nv] = wmma_bf(a, b, acc[nv]);
    }
  }
  int n = lane & 15, mro = (lane >> 4) * 8;
#pragma unroll
  for (int nv = 0; nv < 11; ++nv) {
    int col = nv * 16 + n;
    if (col >= NCOUT) continue;
    float bl = gldf(blin + col);
#pragma unroll
    for (int r = 0; r < 8; ++r) {
      int m = mrow0 + mro + r;
      gstf(out + (size_t)m * NCOUT + col, sigm(acc[nv][r] + bl));
    }
  }
}

__global__ __launch_bounds__(256)
void k_scan(PArgs P) {
  const int lane = threadIdx.x & 31;
  const int wid = blockIdx.x * 8 + (threadIdx.x >> 5);
  const int TOTW = NBLK * 8;
  u32 eloc = 0;
  u16*   hbuf[2]  = { P.h0,  P.h1  };
  u16*   hhbuf[2] = { P.hh0, P.hh1 };
  float* cbuf[2]  = { P.c0,  P.c1  };
  float* hcbuf[2] = { P.hc0, P.hc1 };

  for (int l = 0; l < 2; ++l) {
    const LayerW W = P.L[l];
    {  // zero the "cur" recurrent state
      int tid = blockIdx.x * 256 + threadIdx.x;
      int stride = NBLK * 256;
      u32* z0 = (u32*)hbuf[0];
      u32* z1 = (u32*)hhbuf[0];
      u32* z2 = (u32*)cbuf[0];
      u32* z3 = (u32*)hcbuf[0];
      for (int i = tid; i < BATCH * DH / 2;  i += stride) gst32(z0 + i, 0u);
      for (int i = tid; i < BATCH * DHH / 2; i += stride) gst32(z1 + i, 0u);
      for (int i = tid; i < BATCH * DH;      i += stride) gst32(z2 + i, 0u);
      for (int i = tid; i < BATCH * DHH;     i += stride) gst32(z3 + i, 0u);
    }
    gbar(P.cnt, P.epoch, eloc, NBLK);
    int cur = 0;
    for (int t = 0; t < TT; ++t) {
      int nxt = cur ^ 1;
      for (int task = wid; task < 288; task += TOTW)
        stage1_task(task, lane, W, t, hbuf[cur], hhbuf[cur], hcbuf[cur],
                    hcbuf[nxt], hhbuf[nxt], P.ax, P.ah);
      gbar(P.cnt, P.epoch, eloc, NBLK);
      for (int task = wid; task < 192; task += TOTW)
        stage2_task(task, lane, W, hhbuf[nxt], P.zbuf);
      gbar(P.cnt, P.epoch, eloc, NBLK);
      for (int task = wid; task < 64; task += TOTW)
        stage3_task(task, lane, W, P.zbuf, P.ax, P.ah, cbuf[cur], cbuf[nxt],
                    hbuf[nxt], t);
      gbar(P.cnt, P.epoch, eloc, NBLK);
      cur = nxt;
    }
  }
  // final linear over all (b,t) rows of layer-1 output
  for (int task = wid; task < 800; task += TOTW)
    final_task(task, lane, P.L[1].hstore, P.wlinp, P.blin, P.out);
}

// ---------------- host launcher ----------------

extern "C" void kernel_launch(void* const* d_in, const int* in_sizes, int n_in,
                              void* d_out, int out_size, void* d_ws, size_t ws_size,
                              hipStream_t stream) {
  (void)in_sizes; (void)n_in; (void)out_size; (void)ws_size;
  const float* x = (const float*)d_in[0];
  const float* Pp[2][14];
  for (int l = 0; l < 2; ++l)
    for (int i = 0; i < 14; ++i)
      Pp[l][i] = (const float*)d_in[1 + 14 * l + i];
  const float* Wlin = (const float*)d_in[29];
  const float* blin = (const float*)d_in[30];
  float* out = (float*)d_out;

  // param indices: 0 Wx, 1 Wh, 2 b0, 3 hWx, 4 hWh, 5 hb, 6 Wzx, 7 Wzh, 8 Wzb,
  //                9 bzx, 10 bzh, 11 Wdx, 12 Wdh, 13 Wdb
  size_t off = 0;
  auto alloc = [&](size_t bytes) -> void* {
    void* p = (char*)d_ws + off;
    off += (bytes + 255) & ~(size_t)255;
    return p;
  };
  u16* xbf = (u16*)alloc((size_t)BATCH * TT * DINP * 2);
  u16* h1s = (u16*)alloc((size_t)BATCH * TT * DH * 2);
  u16* h2s = (u16*)alloc((size_t)BATCH * TT * DH * 2);
  u16 *whg[2], *wxp[2], *whp[2], *wz[2], *wd[2];
  float* zbias[2];
  for (int l = 0; l < 2; ++l) {
    const int d_pad = l ? DH : DINP;
    whg[l] = (u16*)alloc((size_t)(d_pad + DH + DHH) * 2048 * 2);
    wxp[l] = (u16*)alloc((size_t)d_pad * 4096 * 2);
    whp[l] = (u16*)alloc((size_t)DH * 4096 * 2);
    wz[l]  = (u16*)alloc((size_t)512 * 6144 * 2);
    wd[l]  = (u16*)alloc((size_t)12 * 512 * 1024 * 2);
    zbias[l] = (float*)alloc(6144 * 4);
  }
  u16* wlinp = (u16*)alloc((size_t)176 * 1024 * 2);
  u16* hb0 = (u16*)alloc(BATCH * DH * 2);
  u16* hb1 = (u16*)alloc(BATCH * DH * 2);
  u16* hhb0 = (u16*)alloc(BATCH * DHH * 2);
  u16* hhb1 = (u16*)alloc(BATCH * DHH * 2);
  float* cb0 = (float*)alloc(BATCH * DH * 4);
  float* cb1 = (float*)alloc(BATCH * DH * 4);
  float* hcb0 = (float*)alloc(BATCH * DHH * 4);
  float* hcb1 = (float*)alloc(BATCH * DHH * 4);
  float* ax = (float*)alloc((size_t)BATCH * 4096 * 4);
  float* ah = (float*)alloc((size_t)BATCH * 4096 * 4);
  u16* zbuf = (u16*)alloc((size_t)BATCH * 6144 * 2);
  u32* barmem = (u32*)alloc(2 * 4);   // [0]=count, [1]=epoch

  // input conversion, Wlin pack, barrier init
  {
    int total = BATCH * TT * DINP;
    k_cvt_x<<<(total + 255) / 256, 256, 0, stream>>>(x, xbf, total);
    k_pack_b<<<32 * 11, 32, 0, stream>>>(Wlin, NCOUT, 0, 0, 1024, NCOUT,
                                         wlinp, 0, 0, 32, 32);
    k_zero<<<1, 256, 0, stream>>>(barmem, 2);
  }

  // pack per-layer weights (bf16, WMMA B-tile layout; L2-resident afterwards)
  for (int l = 0; l < 2; ++l) {
    const int d_in_l = l ? DH : DIN;
    const int d_pad  = l ? DH : DINP;
    const int kh = (d_pad + DH + DHH) / 32;   // 80 or 54
    const int kx = d_pad / 32;                // 32 or 6
    const float* const* p = Pp[l];

    // Whg = rows [hWx_x | pad | hWx_h | hWh], N = 2048
    k_pack_b<<<kx * 128, 32, 0, stream>>>(p[3], 2048, 0, 0, d_in_l, 2048,
                                          whg[l], 0, 0, kh, kx);
    k_pack_b<<<32 * 128, 32, 0, stream>>>(p[3], 2048, d_in_l, 0, 1024, 2048,
                                          whg[l], kx, 0, kh, 32);
    k_pack_b<<<16 * 128, 32, 0, stream>>>(p[4], 2048, 0, 0, 512, 2048,
                                          whg[l], kx + 32, 0, kh, 16);
    // Wx (rows padded), Wh
    k_pack_b<<<kx * 256, 32, 0, stream>>>(p[0], 4096, 0, 0, d_in_l, 4096,
                                          wxp[l], 0, 0, kx, kx);
    k_pack_b<<<32 * 256, 32, 0, stream>>>(p[1], 4096, 0, 0, 1024, 4096,
                                          whp[l], 0, 0, 32, 32);
    // Wz = [Wzx | Wzh | Wzb] column-concat, N = 6144
    for (int s = 0; s < 3; ++s)
      k_pack_b<<<16 * 128, 32, 0, stream>>>(p[6 + s], 2048, 0, 0, 512, 2048,
                                            wz[l], 0, s * 128, 16, 16);
    // Wdx/Wdh/Wdb[g] : 12 x (512 x 1024)
    for (int s = 0; s < 3; ++s)
      for (int g = 0; g < 4; ++g)
        k_pack_b<<<16 * 64, 32, 0, stream>>>(p[11 + s] + (size_t)g * 512 * 1024,
                                             1024, 0, 0, 512, 1024,
                                             wd[l] + (size_t)(s * 4 + g) * 524288,
                                             0, 0, 16, 16);
    k_zbias<<<24, 256, 0, stream>>>(p[9], p[10], zbias[l]);
  }

  // one persistent kernel runs the whole 2-layer scan + final linear
  PArgs P;
  for (int l = 0; l < 2; ++l) {
    const int d_pad = l ? DH : DINP;
    P.L[l].xsrc = l ? h1s : xbf;
    P.L[l].whg = whg[l]; P.L[l].wxp = wxp[l]; P.L[l].whp = whp[l];
    P.L[l].wz = wz[l];   P.L[l].wd = wd[l];
    P.L[l].hb = Pp[l][5]; P.L[l].b0 = Pp[l][2]; P.L[l].zbias = zbias[l];
    P.L[l].hstore = l ? h2s : h1s;
    P.L[l].d_pad = d_pad;
    P.L[l].kh = (d_pad + DH + DHH) / 32;
    P.L[l].kx = d_pad / 32;
    P.L[l].xrs = TT * d_pad;
  }
  P.h0 = hb0; P.h1 = hb1; P.hh0 = hhb0; P.hh1 = hhb1;
  P.c0 = cb0; P.c1 = cb1; P.hc0 = hcb0; P.hc1 = hcb1;
  P.ax = ax; P.ah = ah; P.zbuf = zbuf;
  P.wlinp = wlinp; P.blin = blin; P.out = out;
  P.cnt = barmem; P.epoch = barmem + 1;

  k_scan<<<NBLK, 256, 0, stream>>>(P);
}
